// RNN_seplay_selinh_86148454023841
// MI455X (gfx1250) — compile-verified, matrix operand order checked by reading, not run on hardware
//
#include <hip/hip_runtime.h>

// ---------------- problem constants (from reference) ----------------
#define HE      1024
#define HI      256
#define NH      1280          // HE + HI combined hidden
#define NB      64            // batch
#define TT      1000          // time steps incl. initial
#define NSTEPS  999           // recurrence steps
#define GSZE    307           // int(0.3*1024+0.5)
#define GSZI    77            // int(0.3*256 +0.5)
#define ALPHA_F 0.1f
#define NOISE_SCALE 0.06708203932499369f   // sqrt(2*0.1*0.15^2)

#define NWG     20            // 1280 output neurons / 64 per workgroup
#define WGSIZE  256           // 8 wave32

typedef __attribute__((ext_vector_type(16))) __bf16        v16bf;
typedef __attribute__((ext_vector_type(8)))  float         v8f;
typedef __attribute__((ext_vector_type(4)))  unsigned int  v4u;
typedef __attribute__((ext_vector_type(8)))  unsigned int  v8u;

__device__ __forceinline__ unsigned short f32_to_bf16_rne(float f) {
  unsigned u = __builtin_bit_cast(unsigned, f);
  u += 0x7FFFu + ((u >> 16) & 1u);            // round-to-nearest-even
  return (unsigned short)(u >> 16);
}

__device__ __forceinline__ int grp(int x, int g) {
  return x < g ? 0 : (x < 2 * g ? 1 : -1);
}

// Load a 16x16x32 bf16 WMMA operand fragment (A or B role, same layout):
// lane already selected row/col and +8 K-offset for upper half-wave; this
// reads K elements {0..7} and {16..23} relative to p (two 16B chunks).
__device__ __forceinline__ v16bf load_frag(const unsigned short* p) {
  v4u lo = *(const v4u*)(p);
  v4u hi = *(const v4u*)(p + 16);
  v8u w;
  w[0] = lo[0]; w[1] = lo[1]; w[2] = lo[2]; w[3] = lo[3];
  w[4] = hi[0]; w[5] = hi[1]; w[6] = hi[2]; w[7] = hi[3];
  return __builtin_bit_cast(v16bf, w);
}

// ---------------- kernel 1: build combined bf16 weight matrix ----------------
// Wt[n][k] (n = output neuron 0..1279, k = presynaptic 0..1279), W^T layout.
// With NS=2 the sigma normalizer ns/(ns+s*(2-ns)) == 1 exactly.
__global__ void prep_weights(const float* wee0p, const float* wei0p,
                             const float* wie0p, const float* wii0p,
                             const float* seep,  const float* seip,
                             const float* siep,  const float* siip,
                             unsigned short* __restrict__ Wt) {
  int idx = blockIdx.x * blockDim.x + threadIdx.x;
  if (idx >= NH * NH) return;
  int n = idx / NH, k = idx % NH;
  float wee0 = *wee0p, wei0 = *wei0p, wie0 = *wie0p, wii0 = *wii0p;
  float see = *seep, sei = *seip, sie = *siep, sii = *siip;
  float v;
  if (n < HE) {
    if (k < HE) {                               // +W_ee
      int gr = grp(n, GSZE), gc = grp(k, GSZE);
      v = (gr >= 0 && gc >= 0) ? wee0 * ((gr == gc) ? (1.f + see) : (1.f - see)) : wee0;
    } else {                                    // -W_ie
      int gr = grp(n, GSZE), gc = grp(k - HE, GSZI);
      v = (gr >= 0 && gc >= 0) ? wie0 * ((gr == gc) ? (1.f + sie) : (1.f - sie)) : wie0;
      v = -v;
    }
  } else {
    int ni = n - HE;
    if (k < HE) {                               // +W_ei (block uses wie0 per reference)
      int gr = grp(ni, GSZI), gc = grp(k, GSZE);
      v = (gr >= 0 && gc >= 0) ? wie0 * ((gr == gc) ? (1.f + sei) : (1.f - sei)) : wei0;
    } else {                                    // -W_ii
      int gr = grp(ni, GSZI), gc = grp(k - HE, GSZI);
      v = (gr >= 0 && gc >= 0) ? wii0 * ((gr == gc) ? (1.f + sii) : (1.f - sii)) : wii0;
      v = -v;
    }
  }
  Wt[idx] = f32_to_bf16_rne(v);
}

// ---------------- kernel 2: init state buffers + t=0 outputs + sync reset ----
__global__ void init_state(const float* __restrict__ s_in0,
                           const float* __restrict__ s_e0,
                           const float* __restrict__ s_i0,
                           float* __restrict__ hbuf0,
                           unsigned short* __restrict__ rbuf0,
                           float* __restrict__ sinbuf0,
                           float* __restrict__ out_states_in,
                           float* __restrict__ out_act_e,
                           float* __restrict__ out_act_i,
                           int* __restrict__ sync) {
  int idx = blockIdx.x * blockDim.x + threadIdx.x;
  if (idx < NB * NH) {
    int b = idx / NH, n = idx % NH;
    float h = (n < HE) ? s_e0[b * HE + n] : s_i0[b * HI + (n - HE)];
    float r = fmaxf(h, 0.f);
    hbuf0[idx] = h;
    rbuf0[idx] = f32_to_bf16_rne(r);
    if (n < HE) out_act_e[(size_t)b * TT * HE + n] = r;
    else        out_act_i[(size_t)b * TT * HI + (n - HE)] = r;
  }
  if (idx < NB * 2) {
    int b = idx >> 1, c = idx & 1;
    float s = s_in0[b * 2 + c];
    sinbuf0[idx] = s;
    out_states_in[(size_t)b * TT * 2 + c] = s;
  }
  if (idx < 4) sync[idx] = 0;
}

// ---------------- kernel 3: persistent recurrence -----------------
// Grid = 20 WGs; WG `wg` owns output neurons [64*wg, 64*wg+64). Its 1280x64
// bf16 W^T slice (160 KB) lives in LDS for all 999 steps. Per step: bf16 WMMA
// GEMM (M=64 cols, N=64 batch, K=1280), f32 epilogue, device-wide barrier.
__global__ void __launch_bounds__(WGSIZE, 1)
rnn_persistent(const unsigned short* __restrict__ Wt,
               const float* __restrict__ u,
               const float* __restrict__ noise_e,
               const float* __restrict__ noise_i,
               const float* __restrict__ win0p,
               float* __restrict__ hbuf,            // [2][NB*NH] f32 state
               unsigned short* __restrict__ rbuf,   // [2][NB*NH] bf16 relu(state)
               float* __restrict__ sinbuf,          // [2][NB*2]
               float* __restrict__ out_states_in,
               float* __restrict__ out_act_e,
               float* __restrict__ out_act_i,
               int* __restrict__ sync_count,
               int* __restrict__ sync_gen) {
  __shared__ __align__(16) unsigned short Wlds[64 * NH];   // 160 KB of 320 KB WGP LDS

  const int wg   = blockIdx.x;       // 0..19
  const int n0   = wg * 64;          // first output neuron of this WG
  const int tid  = threadIdx.x;
  const int wave = tid >> 5;
  const int lane = tid & 31;
  const int l16  = lane & 15;
  const int lhalf = lane >> 4;       // upper half-wave handles K+8
  const float win0 = *win0p;
  const float a = ALPHA_F, one_m_a = 1.f - ALPHA_F;

  // one-time weight slice fill: global -> LDS, 16B per thread per iter
  {
    const v4u* src = (const v4u*)(Wt + (size_t)n0 * NH);
    v4u* dst = (v4u*)Wlds;
    for (int i = tid; i < (64 * NH * 2) / 16; i += WGSIZE) dst[i] = src[i];
  }
  __syncthreads();

  // tile assignment: wave -> (M subtile mi, two N subtiles ni0, ni0+1)
  const int mi  = wave & 3;            // output-col subtile (A matrix rows)
  const int ni0 = (wave >> 2) * 2;     // batch subtiles (B matrix cols)

  // A fragment base: weight row = mi*16 + l16 in LDS, K chunk +8 for upper half
  const unsigned short* arow = Wlds + (size_t)(mi * 16 + l16) * NH + lhalf * 8;

  int cur = 0;
  for (int t = 0; t < NSTEPS; ++t) {
    const unsigned short* rb = rbuf + (size_t)cur * (NB * NH);
    const unsigned short* b0row = rb + (size_t)(ni0 * 16 + l16) * NH + lhalf * 8;
    const unsigned short* b1row = rb + (size_t)((ni0 + 1) * 16 + l16) * NH + lhalf * 8;

    v8f acc0 = {}; v8f acc1 = {};
    #pragma unroll 4
    for (int kb = 0; kb < NH / 32; ++kb) {
      const int ko = kb * 32;
      v16bf af  = load_frag(arow  + ko);   // weights from LDS (reused for both N tiles)
      v16bf bf0 = load_frag(b0row + ko);   // activations from L2-resident ping-pong
      v16bf bf1 = load_frag(b1row + ko);
      acc0 = __builtin_amdgcn_wmma_f32_16x16x32_bf16(false, af, false, bf0,
                                                     (short)0, acc0, false, false);
      acc1 = __builtin_amdgcn_wmma_f32_16x16x32_bf16(false, af, false, bf1,
                                                     (short)0, acc1, false, false);
    }

    // ---- epilogue: leak + input current + noise, write state & outputs ----
    const float* hc = hbuf + (size_t)cur * (NB * NH);
    float* hn = hbuf + (size_t)(cur ^ 1) * (NB * NH);
    unsigned short* rn = rbuf + (size_t)(cur ^ 1) * (NB * NH);
    const float* sinc = sinbuf + (size_t)cur * (NB * 2);
    const int nbase = n0 + mi * 16 + lhalf * 8;      // 8 consecutive output neurons

    #pragma unroll
    for (int which = 0; which < 2; ++which) {
      v8f acc = which ? acc1 : acc0;
      const int b = (ni0 + which) * 16 + l16;        // batch row (D-matrix N = lane)
      const float s0 = sinc[b * 2 + 0] * win0;
      const float s1 = sinc[b * 2 + 1] * win0;
      #pragma unroll
      for (int r = 0; r < 8; ++r) {
        const int n = nbase + r;
        const int g = (n < HE) ? grp(n, GSZE) : grp(n - HE, GSZI);
        const float cur_in = (g == 0) ? s0 : ((g == 1) ? s1 : 0.f);
        const float nz = (n < HE)
            ? noise_e[((size_t)b * TT + t) * HE + n]
            : noise_i[((size_t)b * TT + t) * HI + (n - HE)];
        const float hold = hc[(size_t)b * NH + n];
        const float hnew = one_m_a * hold + a * (acc[r] + cur_in) + NOISE_SCALE * nz;
        const float relu = fmaxf(hnew, 0.f);
        hn[(size_t)b * NH + n] = hnew;
        rn[(size_t)b * NH + n] = f32_to_bf16_rne(relu);
        if (n < HE) out_act_e[((size_t)b * TT + (t + 1)) * HE + n] = relu;
        else        out_act_i[((size_t)b * TT + (t + 1)) * HI + (n - HE)] = relu;
      }
    }

    // s_in recurrence (tiny): WG 0 only; uses OLD s_in for input current above
    if (wg == 0 && tid < NB * 2) {
      const int b = tid >> 1, c = tid & 1;
      const float sold = sinbuf[(size_t)cur * (NB * 2) + tid];
      const float snew = one_m_a * sold + a * u[((size_t)b * TT + t) * 2 + c];
      sinbuf[(size_t)(cur ^ 1) * (NB * 2) + tid] = snew;
      out_states_in[((size_t)b * TT + (t + 1)) * 2 + c] = snew;
    }

    // ---- device-wide sense-reversing barrier (20 co-resident WGs) ----
    __syncthreads();
    if (tid == 0) {
      __threadfence();
      int g = __hip_atomic_load(sync_gen, __ATOMIC_ACQUIRE, __HIP_MEMORY_SCOPE_AGENT);
      int prev = __hip_atomic_fetch_add(sync_count, 1, __ATOMIC_ACQ_REL,
                                        __HIP_MEMORY_SCOPE_AGENT);
      if (prev == NWG - 1) {
        __hip_atomic_store(sync_count, 0, __ATOMIC_RELAXED, __HIP_MEMORY_SCOPE_AGENT);
        __hip_atomic_store(sync_gen, g + 1, __ATOMIC_RELEASE, __HIP_MEMORY_SCOPE_AGENT);
      } else {
        while (__hip_atomic_load(sync_gen, __ATOMIC_ACQUIRE,
                                 __HIP_MEMORY_SCOPE_AGENT) == g)
          __builtin_amdgcn_s_sleep(2);
      }
    }
    __syncthreads();
    cur ^= 1;
  }
}

// ---------------- kernel 4: states_o = act_e @ (wou0*w_out)^T ---------------
// Block per (b,t); threads 0..127 sum cols [0,307), 128..255 sum [307,614).
__global__ void compute_states_o(const float* __restrict__ act_e,
                                 const float* __restrict__ wou0p,
                                 float* __restrict__ out_o) {
  const int b = blockIdx.x / TT, tt = blockIdx.x % TT;
  const float* row = act_e + ((size_t)b * TT + tt) * HE;
  __shared__ float red[WGSIZE];
  const int tid = threadIdx.x;
  const int gidx = tid >> 7;         // 0 or 1
  const int lt = tid & 127;
  float s = 0.f;
  for (int c = lt; c < GSZE; c += 128) s += row[gidx * GSZE + c];
  red[tid] = s;
  __syncthreads();
  for (int off = 64; off > 0; off >>= 1) {
    if (lt < off) red[tid] += red[tid + off];
    __syncthreads();
  }
  if (lt == 0) out_o[((size_t)b * TT + tt) * 2 + gidx] = red[tid] * (*wou0p);
}

// ---------------- launch ----------------
extern "C" void kernel_launch(void* const* d_in, const int* in_sizes, int n_in,
                              void* d_out, int out_size, void* d_ws, size_t ws_size,
                              hipStream_t stream) {
  (void)in_sizes; (void)n_in; (void)out_size; (void)ws_size;
  const float* u       = (const float*)d_in[0];
  const float* s_in0   = (const float*)d_in[1];
  const float* s_e0    = (const float*)d_in[2];
  const float* s_i0    = (const float*)d_in[3];
  const float* noise_e = (const float*)d_in[4];
  const float* noise_i = (const float*)d_in[5];
  const float* wee0 = (const float*)d_in[6];
  const float* wei0 = (const float*)d_in[7];
  const float* wie0 = (const float*)d_in[8];
  const float* wii0 = (const float*)d_in[9];
  const float* win0 = (const float*)d_in[10];
  const float* wou0 = (const float*)d_in[11];
  const float* see  = (const float*)d_in[12];
  const float* sei  = (const float*)d_in[13];
  const float* sie  = (const float*)d_in[14];
  const float* sii  = (const float*)d_in[15];

  float* out     = (float*)d_out;
  float* out_o   = out;                 // (B,T,2)   = 128000
  float* out_sin = out + 128000;        // (B,T,2)   = 128000
  float* out_ae  = out + 256000;        // (B,T,HE)  = 65,536,000
  float* out_ai  = out + 65792000;      // (B,T,HI)  = 16,384,000

  // workspace carve (all chunks 16B-aligned)
  char* ws = (char*)d_ws;
  unsigned short* Wt = (unsigned short*)ws;  ws += (size_t)NH * NH * 2;      // 3.28 MB
  unsigned short* rbuf = (unsigned short*)ws; ws += (size_t)2 * NB * NH * 2; // 320 KB
  float* hbuf = (float*)ws;                  ws += (size_t)2 * NB * NH * 4;  // 640 KB
  float* sinbuf = (float*)ws;                ws += (size_t)2 * NB * 2 * 4;   // 1 KB
  int* sync = (int*)ws;                                                     // count, gen

  prep_weights<<<(NH * NH + 255) / 256, 256, 0, stream>>>(
      wee0, wei0, wie0, wii0, see, sei, sie, sii, Wt);
  init_state<<<(NB * NH + 255) / 256, 256, 0, stream>>>(
      s_in0, s_e0, s_i0, hbuf, rbuf, sinbuf, out_sin, out_ae, out_ai, sync);
  rnn_persistent<<<NWG, WGSIZE, 0, stream>>>(
      Wt, u, noise_e, noise_i, win0, hbuf, rbuf, sinbuf,
      out_sin, out_ae, out_ai, &sync[0], &sync[1]);
  compute_states_o<<<NB * TT, WGSIZE, 0, stream>>>(out_ae, wou0, out_o);
}
